// AdvancedHeteroLinkPredictor_75591424409997
// MI455X (gfx1250) — matrix operand order repaired and checked by reading, not compiled, not run on hardware
//
#include <hip/hip_runtime.h>

// ---------------------------------------------------------------------------
// MI455X (gfx1250) implementation of AdvancedHeteroLinkPredictor.
// ~120 GF of GEMM work (bf16 WMMA, f32 accumulate) vs ~5-8 GB of random edge
// gather/scatter traffic + ~1G f32 atomics -> memory/atomic bound overall.
// This revision stages the packed weight matrices in LDS (one coalesced copy
// per block + barrier); the WMMA K-chain then feeds from ds_load_b128 instead
// of stalling on an L2-latency global load before every v_wmma.
// ---------------------------------------------------------------------------

typedef __attribute__((ext_vector_type(16))) __bf16           v16bf;
typedef __attribute__((ext_vector_type(16))) unsigned short   v16us;
typedef __attribute__((ext_vector_type(8)))  float            v8f;

#define HH 128

__device__ __forceinline__ unsigned short f2bf(float x) {
    union { float f; unsigned u; } v; v.f = x;
    unsigned r = v.u + 0x7FFFu + ((v.u >> 16) & 1u);   // RNE
    return (unsigned short)(r >> 16);
}
__device__ __forceinline__ unsigned pack2bf(float lo, float hi) {
    return (unsigned)f2bf(lo) | ((unsigned)f2bf(hi) << 16);
}
__device__ __forceinline__ float bf2f(unsigned short h) {
    union { unsigned u; float f; } v; v.u = ((unsigned)h) << 16;
    return v.f;
}

// A-fragment (16x32 bf16, MxK) from row-major bf16 [M, ldk].
// Lane L: m = m0 + (L&15), half h = L>>4; elems e0..7 = K[k0+8h .. +7],
// e8..15 = K[k0+16+8h .. +7]  (ISA 7.12.2, 16-bit A layout).
__device__ __forceinline__ v16us load_afrag(const unsigned short* __restrict__ A,
                                            int row, int ldk, int k0, int h) {
    const uint4* p0 = (const uint4*)(A + (long)row * ldk + k0 + 8 * h);
    const uint4* p1 = (const uint4*)(A + (long)row * ldk + k0 + 16 + 8 * h);
    union { uint4 q[2]; v16us v; } u;
    u.q[0] = p0[0]; u.q[1] = p1[0];
    return u.v;
}

// B-fragment from packed weights (LDS or global): base[((kt*NT+nt)*32+lane)*16 + e]
__device__ __forceinline__ v16us load_bfrag(const unsigned short* Wp,
                                            int kt, int nt, int NT, int lane) {
    const uint4* p = (const uint4*)(Wp + ((((kt * NT + nt) << 5) + lane) << 4));
    union { uint4 q[2]; v16us v; } u;
    u.q[0] = p[0]; u.q[1] = p[1];
    return u.v;
}

// Coalesced block-wide copy of packed weights into LDS (uint4 granularity).
__device__ __forceinline__ void stage_w_lds(unsigned short* sW,
                                            const unsigned short* __restrict__ gW,
                                            int totalHalves) {
    for (int i = threadIdx.x * 8; i < totalHalves; i += blockDim.x * 8)
        *(uint4*)(sW + i) = *(const uint4*)(gW + i);
}

__device__ __forceinline__ v8f wmma_bf16(v16us a, v16us b, v8f c) {
    return __builtin_amdgcn_wmma_f32_16x16x32_bf16(
        false, __builtin_bit_cast(v16bf, a),
        false, __builtin_bit_cast(v16bf, b),
        (short)0, c, false, false);
}

// ---------------------------------------------------------------------------
// Utility kernels
// ---------------------------------------------------------------------------
__global__ void fill_f32(float* __restrict__ p, float v, long n) {
    long i = ((long)blockIdx.x * blockDim.x + threadIdx.x) * 4;
    if (i + 3 < n) {
        float4 q = {v, v, v, v};
        *(float4*)(p + i) = q;
    } else {
        for (long j = i; j < n; ++j) p[j] = v;
    }
}

__global__ void count_deg(const int* __restrict__ idx, float* __restrict__ cnt, int nE) {
    int i = blockIdx.x * blockDim.x + threadIdx.x;
    if (i < nE) unsafeAtomicAdd(&cnt[idx[i]], 1.0f);
}

__global__ void rsqrtize(float* __restrict__ p, long n) {
    long i = (long)blockIdx.x * blockDim.x + threadIdx.x;
    if (i < n) p[i] = rsqrtf(fmaxf(p[i], 1.0f));
}

// dst_bf16[i..i+7] = bf16(src[i..] * rs[row]); 8 elems/thread, never straddles
// a 128-wide row since i % 8 == 0.
__global__ void scale_cvt(unsigned short* __restrict__ dst, const float* __restrict__ src,
                          const float* __restrict__ rs, long n) {
    long i = ((long)blockIdx.x * blockDim.x + threadIdx.x) * 8;
    if (i >= n) return;
    float w = rs ? rs[i >> 7] : 1.0f;
    const float4* p = (const float4*)(src + i);
    float4 q0 = p[0], q1 = p[1];
    uint4 o;
    o.x = pack2bf(q0.x * w, q0.y * w);
    o.y = pack2bf(q0.z * w, q0.w * w);
    o.z = pack2bf(q1.x * w, q1.y * w);
    o.w = pack2bf(q1.z * w, q1.w * w);
    *(uint4*)(dst + i) = o;
}

// Pack row-major W[K,N] (f32, optional per-row scale) into B-fragment order.
__global__ void pack_w(const float* __restrict__ W, unsigned short* __restrict__ out,
                       int K, int N, const float* __restrict__ rowscale) {
    int i = blockIdx.x * blockDim.x + threadIdx.x;
    if (i >= K * N) return;
    int kk = i / N, nn = i - kk * N;
    int NT = N >> 4;
    int kt = kk >> 5, k32 = kk & 31, nt = nn >> 4, n = nn & 15;
    int lane = n + 16 * ((k32 >> 3) & 1);
    int e = (k32 & 7) + 8 * (k32 >> 4);
    float v = W[i];
    if (rowscale) v *= rowscale[kk];
    out[((((kt * NT + nt) << 5) + lane) << 4) + e] = f2bf(v);
}

// h[i..i+3] = maybe_relu( (agg + bA + bB) * s ), 4 elems/thread (f % 4 == 0)
__global__ void finalize_nodes(float* __restrict__ h, const float* __restrict__ agg,
                               const float* __restrict__ bA, const float* __restrict__ bB,
                               float s, int do_relu, long n) {
    long i = ((long)blockIdx.x * blockDim.x + threadIdx.x) * 4;
    if (i >= n) return;
    int f = (int)(i & 127);
    float4 a = *(const float4*)(agg + i);
    float4 ba = *(const float4*)(bA + f);
    float4 bb = bB ? *(const float4*)(bB + f) : float4{0.f, 0.f, 0.f, 0.f};
    float4 r;
    r.x = (a.x + ba.x + bb.x) * s;
    r.y = (a.y + ba.y + bb.y) * s;
    r.z = (a.z + ba.z + bb.z) * s;
    r.w = (a.w + ba.w + bb.w) * s;
    if (do_relu) {
        r.x = fmaxf(r.x, 0.f); r.y = fmaxf(r.y, 0.f);
        r.z = fmaxf(r.z, 0.f); r.w = fmaxf(r.w, 0.f);
    }
    *(float4*)(h + i) = r;
}

// agg[dst] += t_bf16[src] * rs_in[dst]  (8 threads / edge, 16 feats each)
__global__ void __launch_bounds__(256)
scatter_edges(const unsigned short* __restrict__ t, const int* __restrict__ src,
              const int* __restrict__ dst, const float* __restrict__ rsd,
              float* __restrict__ agg, int nE) {
    long g = (long)blockIdx.x * blockDim.x + threadIdx.x;
    int e = (int)(g >> 3);
    if (e >= nE) return;
    int part = (int)(g & 7);
    int s = src[e], d = dst[e];
    float w = rsd[d];
    const uint4* p = (const uint4*)(t + (long)s * HH + part * 16);
    uint4 q0 = p[0], q1 = p[1];
    unsigned v[8] = {q0.x, q0.y, q0.z, q0.w, q1.x, q1.y, q1.z, q1.w};
    float* ag = agg + (long)d * HH + part * 16;
#pragma unroll
    for (int j = 0; j < 8; ++j) {
        unsafeAtomicAdd(&ag[2 * j],     bf2f((unsigned short)(v[j] & 0xffffu)) * w);
        unsafeAtomicAdd(&ag[2 * j + 1], bf2f((unsigned short)(v[j] >> 16)) * w);
    }
}

// ---------------------------------------------------------------------------
// Generic WMMA GEMM: out[M, NT*16] = A_bf16[M, KT*32] @ Wpack (+bias)
// One wave per 16-row M-tile; 8 waves / block; packed W staged in LDS.
// ---------------------------------------------------------------------------
template <int KT, int NT, bool OUT_BF16>
__global__ void __launch_bounds__(256)
gemm_kernel(const unsigned short* __restrict__ A, const unsigned short* __restrict__ Wp,
            const float* __restrict__ bias, void* __restrict__ out, int Mtiles) {
    __shared__ unsigned short sW[KT * NT * 512];
    stage_w_lds(sW, Wp, KT * NT * 512);
    __syncthreads();

    int wave = blockIdx.x * (blockDim.x >> 5) + (threadIdx.x >> 5);
    if (wave >= Mtiles) return;
    int lane = threadIdx.x & 31;
    int h = lane >> 4, ml = lane & 15;
    int row = wave * 16 + ml;
    v16us a[KT];
#pragma unroll
    for (int kt = 0; kt < KT; ++kt) a[kt] = load_afrag(A, row, KT * 32, kt * 32, h);
#pragma unroll
    for (int nt = 0; nt < NT; ++nt) {
        v8f c = {0.f, 0.f, 0.f, 0.f, 0.f, 0.f, 0.f, 0.f};
#pragma unroll
        for (int kt = 0; kt < KT; ++kt)
            c = wmma_bf16(a[kt], load_bfrag(sW, kt, nt, NT, lane), c);
        int col = nt * 16 + ml;
        float bv = bias ? bias[col] : 0.0f;
#pragma unroll
        for (int r = 0; r < 8; ++r) {
            float vv = c[r] + bv;
            long orow = (long)wave * 16 + r + 8 * h;
            if (OUT_BF16) ((unsigned short*)out)[orow * (NT * 16) + col] = f2bf(vv);
            else          ((float*)out)[orow * (NT * 16) + col] = vv;
        }
    }
}

// ---------------------------------------------------------------------------
// Scoring stage 1: t = relu(concat(hg[src], hd[dst]) @ W_ft + b_ft)
// Fused edge-gather + WMMA (K=256, W_ft in LDS) + per-feature sum/sumsq.
// ---------------------------------------------------------------------------
__global__ void __launch_bounds__(256)
score_stage1(const float* __restrict__ hg, const float* __restrict__ hd,
             const int* __restrict__ src, const int* __restrict__ dst,
             const unsigned short* __restrict__ Wftp, const float* __restrict__ bft,
             unsigned short* __restrict__ tbuf, float* __restrict__ gsum,
             float* __restrict__ gsq, int Etiles) {
    __shared__ unsigned short sW[8 * 8 * 512];   // 64 KB packed W_ft
    __shared__ float s_sum[HH], s_sq[HH];
    int tid = threadIdx.x;
    stage_w_lds(sW, Wftp, 8 * 8 * 512);
    if (tid < HH) { s_sum[tid] = 0.f; s_sq[tid] = 0.f; }
    __syncthreads();

    int wave = blockIdx.x * 8 + (tid >> 5);   // Etiles is a multiple of 8
    int lane = tid & 31;
    int h = lane >> 4, ml = lane & 15;
    int e0 = wave * 16;
    int s = src[e0 + ml], d = dst[e0 + ml];

    v16us a[8];
#pragma unroll
    for (int kt = 0; kt < 8; ++kt) {
        const float* base = (kt < 4) ? (hg + (long)s * HH + kt * 32)
                                     : (hd + (long)d * HH + (kt - 4) * 32);
        union { float4 f4[4]; float f[16]; } u;
        const float4* p0 = (const float4*)(base + 8 * h);
        const float4* p1 = (const float4*)(base + 16 + 8 * h);
        u.f4[0] = p0[0]; u.f4[1] = p0[1];
        u.f4[2] = p1[0]; u.f4[3] = p1[1];
        v16us v;
#pragma unroll
        for (int i = 0; i < 16; ++i) v[i] = f2bf(u.f[i]);
        a[kt] = v;
    }

#pragma unroll
    for (int nt = 0; nt < 8; ++nt) {
        v8f c = {0.f, 0.f, 0.f, 0.f, 0.f, 0.f, 0.f, 0.f};
#pragma unroll
        for (int kt = 0; kt < 8; ++kt)
            c = wmma_bf16(a[kt], load_bfrag(sW, kt, nt, 8, lane), c);
        int col = nt * 16 + ml;
        float bv = bft[col];
        float ps = 0.f, pq = 0.f;
#pragma unroll
        for (int r = 0; r < 8; ++r) {
            float vv = c[r] + bv;
            vv = fmaxf(vv, 0.0f);
            ps += vv; pq += vv * vv;
            tbuf[((long)e0 + r + 8 * h) * HH + col] = f2bf(vv);
        }
        atomicAdd(&s_sum[col], ps);
        atomicAdd(&s_sq[col], pq);
    }
    __syncthreads();
    if (tid < HH) {
        unsafeAtomicAdd(&gsum[tid], s_sum[tid]);
        unsafeAtomicAdd(&gsq[tid], s_sq[tid]);
    }
}

// scale/shift for folded BN + embedding
__global__ void bn_prep(const float* __restrict__ gsum, const float* __restrict__ gsq,
                        const float* __restrict__ gamma, const float* __restrict__ beta,
                        const float* __restrict__ emb, float* __restrict__ scale,
                        float* __restrict__ shift, float invE) {
    int f = threadIdx.x;
    if (f >= HH) return;
    float mu = gsum[f] * invE;
    float var = gsq[f] * invE - mu * mu;
    float inv = rsqrtf(var + 1e-5f);
    float sc = inv * gamma[f];
    scale[f] = sc;
    shift[f] = beta[f] - mu * sc + emb[f];
}

// b'[j] = b_e1[j] + sum_k shift[k] * W_e1[k, j]
__global__ void bias_e1(const float* __restrict__ W_e1, const float* __restrict__ b_e1,
                        const float* __restrict__ shift, float* __restrict__ bp) {
    int j = threadIdx.x;
    if (j >= 64) return;
    float s = b_e1[j];
    for (int k = 0; k < HH; ++k) s += shift[k] * W_e1[k * 64 + j];
    bp[j] = s;
}

// ---------------------------------------------------------------------------
// Scoring stage 2: score = relu(t @ W1' + b') @ W_e2 + b_e2
// (BN affine folded into W1'/b'); W1' staged in LDS; shuffle-reduced K=64 dot.
// ---------------------------------------------------------------------------
__global__ void __launch_bounds__(256)
score_stage2(const unsigned short* __restrict__ tbuf, const unsigned short* __restrict__ We1p,
             const float* __restrict__ bp, const float* __restrict__ w2,
             const float* __restrict__ b2, float* __restrict__ out, int Etiles) {
    __shared__ unsigned short sW[4 * 4 * 512];   // 16 KB packed W1'
    stage_w_lds(sW, We1p, 4 * 4 * 512);
    __syncthreads();

    int wave = blockIdx.x * 8 + (threadIdx.x >> 5);
    if (wave >= Etiles) return;
    int lane = threadIdx.x & 31;
    int h = lane >> 4, ml = lane & 15;
    int row = wave * 16 + ml;
    v16us a[4];
#pragma unroll
    for (int kt = 0; kt < 4; ++kt) a[kt] = load_afrag(tbuf, row, HH, kt * 32, h);

    float acc[8] = {0.f, 0.f, 0.f, 0.f, 0.f, 0.f, 0.f, 0.f};
#pragma unroll
    for (int nt = 0; nt < 4; ++nt) {
        v8f c = {0.f, 0.f, 0.f, 0.f, 0.f, 0.f, 0.f, 0.f};
#pragma unroll
        for (int kt = 0; kt < 4; ++kt)
            c = wmma_bf16(a[kt], load_bfrag(sW, kt, nt, 4, lane), c);
        int col = nt * 16 + ml;
        float bv = bp[col], wv = w2[col];
#pragma unroll
        for (int r = 0; r < 8; ++r) {
            float vv = fmaxf(c[r] + bv, 0.0f);
            acc[r] += vv * wv;
        }
    }
#pragma unroll
    for (int off = 1; off < 16; off <<= 1)
#pragma unroll
        for (int r = 0; r < 8; ++r) acc[r] += __shfl_xor(acc[r], off, 32);
    if (ml == 0) {
        float bb = b2[0];
#pragma unroll
        for (int r = 0; r < 8; ++r) out[wave * 16 + r + 8 * h] = acc[r] + bb;
    }
}

// ---------------------------------------------------------------------------
// Host orchestration
// ---------------------------------------------------------------------------
extern "C" void kernel_launch(void* const* d_in, const int* in_sizes, int n_in,
                              void* d_out, int out_size, void* d_ws, size_t ws_size,
                              hipStream_t stream) {
    (void)in_sizes; (void)n_in; (void)out_size; (void)ws_size;
    const long NG = 200000, ND = 50000;
    const long EGG = 1000000, EGD = 800000, EDG = 800000, ESC = 400000;

    const float* x_gene   = (const float*)d_in[0];
    const float* x_dis    = (const float*)d_in[1];
    const float* Wp_g     = (const float*)d_in[2];
    const float* bp_g     = (const float*)d_in[3];
    const float* Wp_d     = (const float*)d_in[4];
    const float* bp_d     = (const float*)d_in[5];
    const float* W_conv   = (const float*)d_in[6];
    const float* b_conv   = (const float*)d_in[7];
    const float* W_ft     = (const float*)d_in[8];
    const float* b_ft     = (const float*)d_in[9];
    const float* bn_gamma = (const float*)d_in[10];
    const float* bn_beta  = (const float*)d_in[11];
    const float* emb      = (const float*)d_in[12];
    const float* W_e1     = (const float*)d_in[13];
    const float* b_e1     = (const float*)d_in[14];
    const float* W_e2     = (const float*)d_in[15];
    const float* b_e2     = (const float*)d_in[16];
    const int* gg_src = (const int*)d_in[17]; const int* gg_dst = (const int*)d_in[18];
    const int* gd_src = (const int*)d_in[19]; const int* gd_dst = (const int*)d_in[20];
    const int* dg_src = (const int*)d_in[21]; const int* dg_dst = (const int*)d_in[22];
    const int* pos_src = (const int*)d_in[23]; const int* pos_dst = (const int*)d_in[24];
    const int* neg_src = (const int*)d_in[25]; const int* neg_dst = (const int*)d_in[26];

    char* ws = (char*)d_ws;
    size_t off = 0;
    auto alloc = [&](size_t bytes) -> void* {
        void* p = ws + off;
        off = (off + bytes + 255) & ~(size_t)255;
        return p;
    };
    auto cdiv = [](long a, long b) -> int { return (int)((a + b - 1) / b); };

    float* hg    = (float*)alloc(NG * HH * 4);
    float* hd    = (float*)alloc(ND * HH * 4);
    float* agg_g = (float*)alloc(NG * HH * 4);
    float* agg_d = (float*)alloc(ND * HH * 4);
    unsigned short* abuf = (unsigned short*)alloc(NG * HH * 2);
    unsigned short* tbuf = (unsigned short*)alloc(ESC * HH * 2);   // also conv-GEMM temp
    float* rsblk = (float*)alloc((4 * NG + 2 * ND) * 4);
    float* rs_gg_s = rsblk;               // out-deg, gene (gg)
    float* rs_gg_d = rsblk + NG;          // in-deg,  gene (gg)
    float* rs_gd_s = rsblk + 2 * NG;      // out-deg, gene (gd)
    float* rs_dg_d = rsblk + 3 * NG;      // in-deg,  gene (dg)
    float* rs_dg_s = rsblk + 4 * NG;      // out-deg, dis  (dg)
    float* rs_gd_d = rsblk + 4 * NG + ND; // in-deg,  dis  (gd)
    unsigned short* wpk  = (unsigned short*)alloc(11 * 16384 * 2); // 2 proj + 9 conv
    unsigned short* wftp = (unsigned short*)alloc(256 * 128 * 2);
    unsigned short* we1p = (unsigned short*)alloc(128 * 64 * 2);
    float* stats  = (float*)alloc(256 * 4);   // gsum | gsq
    float* gsum = stats, * gsq = stats + 128;
    float* scalev = (float*)alloc(HH * 4);
    float* shiftv = (float*)alloc(HH * 4);
    float* bpbuf  = (float*)alloc(64 * 4);

    // ---- pack all static weights to bf16 B-fragment order ----
    pack_w<<<cdiv(16384, 256), 256, 0, stream>>>(Wp_g, wpk, 128, 128, nullptr);
    pack_w<<<cdiv(16384, 256), 256, 0, stream>>>(Wp_d, wpk + 16384, 128, 128, nullptr);
    for (int i = 0; i < 9; ++i)
        pack_w<<<cdiv(16384, 256), 256, 0, stream>>>(W_conv + (long)i * 16384,
                                                     wpk + (2 + i) * 16384, 128, 128, nullptr);
    pack_w<<<cdiv(256 * 128, 256), 256, 0, stream>>>(W_ft, wftp, 256, 128, nullptr);

    // ---- degree normalizers: rs = rsqrt(max(deg, 1)) ----
    long nrs = 4 * NG + 2 * ND;
    fill_f32<<<cdiv(nrs, 1024), 256, 0, stream>>>(rsblk, 0.f, nrs);
    count_deg<<<cdiv(EGG, 256), 256, 0, stream>>>(gg_src, rs_gg_s, (int)EGG);
    count_deg<<<cdiv(EGG, 256), 256, 0, stream>>>(gg_dst, rs_gg_d, (int)EGG);
    count_deg<<<cdiv(EGD, 256), 256, 0, stream>>>(gd_src, rs_gd_s, (int)EGD);
    count_deg<<<cdiv(EGD, 256), 256, 0, stream>>>(gd_dst, rs_gd_d, (int)EGD);
    count_deg<<<cdiv(EDG, 256), 256, 0, stream>>>(dg_src, rs_dg_s, (int)EDG);
    count_deg<<<cdiv(EDG, 256), 256, 0, stream>>>(dg_dst, rs_dg_d, (int)EDG);
    rsqrtize<<<cdiv(nrs, 256), 256, 0, stream>>>(rsblk, nrs);

    // ---- input projections ----
    scale_cvt<<<cdiv(NG * HH, 2048), 256, 0, stream>>>(abuf, x_gene, nullptr, NG * HH);
    gemm_kernel<4, 8, false><<<cdiv(NG / 16, 8), 256, 0, stream>>>(abuf, wpk, bp_g, hg, (int)(NG / 16));
    scale_cvt<<<cdiv(ND * HH, 2048), 256, 0, stream>>>(abuf, x_dis, nullptr, ND * HH);
    gemm_kernel<4, 8, false><<<cdiv(ND / 16, 8), 256, 0, stream>>>(abuf, wpk + 16384, bp_d, hd, (int)(ND / 16));

    // ---- 3 RGCN layers ----
    for (int l = 0; l < 3; ++l) {
        const float* b0 = b_conv + (l * 3 + 0) * HH;
        const float* b1 = b_conv + (l * 3 + 1) * HH;
        const float* b2 = b_conv + (l * 3 + 2) * HH;
        fill_f32<<<cdiv(NG * HH, 1024), 256, 0, stream>>>(agg_g, 0.f, NG * HH);
        fill_f32<<<cdiv(ND * HH, 1024), 256, 0, stream>>>(agg_d, 0.f, ND * HH);
        // gene->gene
        scale_cvt<<<cdiv(NG * HH, 2048), 256, 0, stream>>>(abuf, hg, rs_gg_s, NG * HH);
        gemm_kernel<4, 8, true><<<cdiv(NG / 16, 8), 256, 0, stream>>>(
            abuf, wpk + (2 + l * 3 + 0) * 16384, nullptr, tbuf, (int)(NG / 16));
        scatter_edges<<<cdiv(EGG * 8, 256), 256, 0, stream>>>(tbuf, gg_src, gg_dst, rs_gg_d, agg_g, (int)EGG);
        // gene->disease
        scale_cvt<<<cdiv(NG * HH, 2048), 256, 0, stream>>>(abuf, hg, rs_gd_s, NG * HH);
        gemm_kernel<4, 8, true><<<cdiv(NG / 16, 8), 256, 0, stream>>>(
            abuf, wpk + (2 + l * 3 + 1) * 16384, nullptr, tbuf, (int)(NG / 16));
        scatter_edges<<<cdiv(EGD * 8, 256), 256, 0, stream>>>(tbuf, gd_src, gd_dst, rs_gd_d, agg_d, (int)EGD);
        // disease->gene
        scale_cvt<<<cdiv(ND * HH, 2048), 256, 0, stream>>>(abuf, hd, rs_dg_s, ND * HH);
        gemm_kernel<4, 8, true><<<cdiv(ND / 16, 8), 256, 0, stream>>>(
            abuf, wpk + (2 + l * 3 + 2) * 16384, nullptr, tbuf, (int)(ND / 16));
        scatter_edges<<<cdiv(EDG * 8, 256), 256, 0, stream>>>(tbuf, dg_src, dg_dst, rs_dg_d, agg_g, (int)EDG);
        // finalize: hg = relu?(0.5*(agg_g + b0 + b2)), hd = relu?(agg_d + b1)
        finalize_nodes<<<cdiv(NG * HH, 1024), 256, 0, stream>>>(hg, agg_g, b0, b2, 0.5f, l < 2, NG * HH);
        finalize_nodes<<<cdiv(ND * HH, 1024), 256, 0, stream>>>(hd, agg_d, b1, nullptr, 1.0f, l < 2, ND * HH);
    }

    // ---- scoring: pos then neg ----
    int Etiles = (int)(ESC / 16);
    for (int s = 0; s < 2; ++s) {
        const int* esrc = s ? neg_src : pos_src;
        const int* edst = s ? neg_dst : pos_dst;
        fill_f32<<<1, 64, 0, stream>>>(stats, 0.f, 256);
        score_stage1<<<Etiles / 8, 256, 0, stream>>>(hg, hd, esrc, edst, wftp, b_ft,
                                                     tbuf, gsum, gsq, Etiles);
        bn_prep<<<1, 128, 0, stream>>>(gsum, gsq, bn_gamma, bn_beta, emb + 128 /*ETYPE=1*/,
                                       scalev, shiftv, 1.0f / (float)ESC);
        pack_w<<<cdiv(128 * 64, 256), 256, 0, stream>>>(W_e1, we1p, 128, 64, scalev);
        bias_e1<<<1, 64, 0, stream>>>(W_e1, b_e1, shiftv, bpbuf);
        score_stage2<<<Etiles / 8, 256, 0, stream>>>(tbuf, we1p, bpbuf, W_e2, b_e2,
                                                     (float*)d_out + s * ESC, Etiles);
    }
}
